// TransolverNeXtCrossAttention_ShareKV_56916906606959
// MI455X (gfx1250) — matrix operand order repaired and checked by reading, not compiled, not run on hardware
//
#include <hip/hip_runtime.h>
#include <hip/hip_bf16.h>
#include <math.h>

// ---------------- problem dims ----------------
constexpr int Bb  = 2;
constexpr int Nn  = 16384;
constexpr int Mm  = 512;
constexpr int Rr  = 8;
constexpr int Dd  = 384;
constexpr int Hh  = 8;
constexpr int DHh = 48;
constexpr int NFf = 8;
constexpr int FHh = 1024;
constexpr int Ll  = 2;
constexpr int SQ  = Mm + Rr;           // 520
constexpr float MINLF = -5.0f, MAXLF = 3.0f;

typedef unsigned short u16;
typedef unsigned int   u32;
typedef __attribute__((ext_vector_type(16))) __bf16 v16bf;
typedef __attribute__((ext_vector_type(8)))  float  v8f;
typedef __attribute__((ext_vector_type(4)))  unsigned int v4u;
typedef __attribute__((ext_vector_type(8)))  int v8i;
typedef __attribute__((ext_vector_type(4)))  int v4i;

// Tensor Data Mover availability (6-arg form: amdgpu-toolchain clang-23)
#if defined(__AMDGCN__) && __has_builtin(__builtin_amdgcn_tensor_load_to_lds) && \
    __has_builtin(__builtin_amdgcn_s_wait_tensorcnt)
#define HAVE_TDM 1
#else
#define HAVE_TDM 0
#endif

// ---------------- scalar helpers ----------------
__device__ __forceinline__ u16 f2bf(float f) {
  u32 u = __builtin_bit_cast(u32, f);
  u32 r = u + 0x7fffu + ((u >> 16) & 1u);   // round-to-nearest-even
  return (u16)(r >> 16);
}
__device__ __forceinline__ float bf2f(u16 h) {
  return __builtin_bit_cast(float, (u32)h << 16);
}

// ---------------- WMMA fragment loaders (layouts per CDNA5 ISA) ----
// A fragment: 16x32 bf16 tile, row-major in LDS, row stride 32 u16 (64B).
// lane l holds row m=l&15; lanes<16 -> K {0..7,16..23}, lanes>=16 -> K {8..15,24..31}
// => two contiguous 16B spans per lane: 2 x ds_load_b128.
union Frag128 { uint4 q[2]; v16bf v; };

__device__ __forceinline__ v16bf load_frag_a(const u16* tile, int lane) {
  Frag128 r;
  const u32* row = (const u32*)(tile + (lane & 15) * 32);
  int kb = (lane >> 4) * 4;
  r.q[0] = *(const uint4*)(row + kb);
  r.q[1] = *(const uint4*)(row + kb + 8);
  return r.v;
}
// B fragment from a COLUMN-MAJOR tile: tile[(n)*32 + k], col stride 32 u16 (64B).
// lane l holds col n=l&15; lanes<16 -> K {0..7,16..23}, lanes>=16 -> K {8..15,24..31}
// => two contiguous 16B spans per lane: 2 x ds_load_b128.
__device__ __forceinline__ v16bf load_frag_b(const u16* tile, int lane) {
  Frag128 r;
  const u32* col = (const u32*)(tile + (lane & 15) * 32);
  int kb = (lane >> 4) * 4;
  r.q[0] = *(const uint4*)(col + kb);
  r.q[1] = *(const uint4*)(col + kb + 8);
  return r.v;
}
__device__ __forceinline__ v8f wmma_bf16(v16bf a, v16bf b, v8f c) {
  return __builtin_amdgcn_wmma_f32_16x16x32_bf16(false, a, false, b, (short)0, c,
                                                 false, false);
}

#if HAVE_TDM
// Issue a TDM 2-D tile load: tileRows x 32 bf16, global row stride ldaEl elements,
// into LDS row-major [row][32]. OOB (beyond validK/validRows) reads return zero.
// Must be executed by a single wave; caller waits tensorcnt then barriers.
__device__ __forceinline__ void tdm_load_tile(u16* lds, const u16* g,
                                              int validK, int validRows,
                                              int ldaEl, int tileRows) {
  unsigned lo = (unsigned)(unsigned long long)(void*)lds;       // LDS byte address
  unsigned long long ga = (unsigned long long)(const void*)g;   // global byte address
  unsigned td0 = (unsigned)(validK  > 0 ? validK  : 0);
  unsigned td1 = (unsigned)(validRows > 0 ? validRows : 0);
  // D# group0: [1:0]=count=1, [63:32]=lds_addr, [120:64]=global_addr, [127:126]=type2
  v4u g0 = { 1u, lo, (unsigned)ga, (unsigned)(ga >> 32) | (2u << 30) };
  // D# group1: data_size=2B; tensor_dim0/1 = valid extent; tile 32 x tileRows;
  // tensor_dim0_stride = ldaEl (data_size units)
  v8i g1 = { (int)(1u << 16),
             (int)((td0 & 0xFFFFu) << 16),
             (int)((td0 >> 16) | ((td1 & 0xFFFFu) << 16)),
             (int)((td1 >> 16) | (32u << 16)),
             (int)(unsigned)tileRows,
             (int)(unsigned)ldaEl,
             0, 0 };
  v4i z4 = { 0, 0, 0, 0 };
  v8i z8 = { 0, 0, 0, 0, 0, 0, 0, 0 };
  __builtin_amdgcn_tensor_load_to_lds(g0, g1, z4, z4, z8, 0);
}
#endif

// ---------------- generic batched bf16 WMMA GEMM ----------------
// C[Mr x Nc] = A[Mr x K] * B + epilogue.  BT==0: B row-major KxN; BT==1: B stored NxK.
// Two-level batch: blockIdx.z = b1*nb2 + b2 (strides sX1 / sX2 in elements).
template <int BT>
__global__ __launch_bounds__(256) void wmma_gemm_kernel(
    const u16* __restrict__ A, long sA1, long sA2, int lda,
    const u16* __restrict__ Bm, long sB1, long sB2, int ldb,
    float* __restrict__ Cf, u16* __restrict__ Cb, long sC1, long sC2, int ldc,
    const float* __restrict__ bias,
    const float* __restrict__ rowscale, long sRS1,
    const float* __restrict__ res,
    int Mr, int Nc, int K, int nb2)
{
  __shared__ alignas(16) u16 ldsA[64 * 32];   // row-major [row][k]
  __shared__ alignas(16) u16 ldsB[64 * 32];   // col-major [n][k]

  const int t    = threadIdx.x;
  const int lane = t & 31;
  const int wave = t >> 5;
  const int mt   = wave >> 1;            // 0..3  (row tile)
  const int nt   = wave & 1;             // 0..1  (col half)

  const int b1 = blockIdx.z / nb2;
  const int b2 = blockIdx.z % nb2;
  const int m0 = blockIdx.x * 64;
  const int n0 = blockIdx.y * 64;

  const u16* Ab = A  + (long)b1 * sA1 + (long)b2 * sA2;
  const u16* Bp = Bm + (long)b1 * sB1 + (long)b2 * sB2;

  v8f acc0 = {}; v8f acc1 = {};

  for (int kk = 0; kk < K; kk += 32) {
    // ---- stage A 64x32 (row-major) ----
#if HAVE_TDM
    if (wave == 0) {
      tdm_load_tile(ldsA, Ab + (long)m0 * lda + kk, K - kk, Mr - m0, lda, 64);
      __builtin_amdgcn_s_wait_tensorcnt(0);
    }
#else
    {
      const int aIdx = t * 8, aRow = aIdx >> 5, aCol = aIdx & 31;
      int gr = m0 + aRow;
      long base = (long)gr * lda + kk + aCol;
      if (gr < Mr && (kk + aCol + 8) <= K) {
        *(uint4*)&ldsA[aIdx] = *(const uint4*)&Ab[base];
      } else {
#pragma unroll
        for (int j = 0; j < 8; ++j)
          ldsA[aIdx + j] = (gr < Mr && (kk + aCol + j) < K) ? Ab[base + j] : (u16)0;
      }
    }
#endif
    // ---- stage B into col-major [n][32] ----
    if (BT == 0) {
      // global row-major KxN: threads t<128 each pack a k-pair for 8 columns
      if (t < 128) {
        int k  = (t >> 3) * 2;           // 0,2,..,30
        int ng = (t & 7) * 8;
        int gk = kk + k, gn = n0 + ng;
        alignas(16) u16 r0[8], r1[8];
        if ((gk + 1) < K && (gn + 8) <= Nc) {
          *(uint4*)r0 = *(const uint4*)&Bp[(long)gk * ldb + gn];
          *(uint4*)r1 = *(const uint4*)&Bp[(long)(gk + 1) * ldb + gn];
        } else {
#pragma unroll
          for (int j = 0; j < 8; ++j) {
            r0[j] = (gk < K && (gn + j) < Nc) ? Bp[(long)gk * ldb + gn + j] : (u16)0;
            r1[j] = ((gk + 1) < K && (gn + j) < Nc) ? Bp[(long)(gk + 1) * ldb + gn + j] : (u16)0;
          }
        }
#pragma unroll
        for (int j = 0; j < 8; ++j)
          *(u32*)&ldsB[(ng + j) * 32 + k] = (u32)r0[j] | ((u32)r1[j] << 16);
        if (kk + 32 < K) __builtin_prefetch(&Bp[(long)(gk + 32) * ldb + gn]);
      }
    } else {
      // global NxK: 8 contiguous k per thread -> contiguous in col-major LDS
      int n  = t >> 2;
      int k0 = (t & 3) * 8;
      int gn = n0 + n, gk = kk + k0;
      alignas(16) u16 tmp[8];
      if (gn < Nc && (gk + 8) <= K) {
        *(uint4*)tmp = *(const uint4*)&Bp[(long)gn * ldb + gk];
      } else {
#pragma unroll
        for (int j = 0; j < 8; ++j)
          tmp[j] = (gn < Nc && (gk + j) < K) ? Bp[(long)gn * ldb + gk + j] : (u16)0;
      }
      *(uint4*)&ldsB[n * 32 + k0] = *(uint4*)tmp;
      if (kk + 32 < K) __builtin_prefetch(&Bp[(long)gn * ldb + gk + 32]);
    }
    __syncthreads();

    v16bf a  = load_frag_a(&ldsA[(mt * 16) * 32], lane);
    v16bf f0 = load_frag_b(&ldsB[(nt * 32) * 32], lane);
    v16bf f1 = load_frag_b(&ldsB[(nt * 32 + 16) * 32], lane);
    acc0 = wmma_bf16(a, f0, acc0);
    acc1 = wmma_bf16(a, f1, acc1);
    __syncthreads();
  }

  // epilogue (C layout: lane<16 rows 0..7, lane>=16 rows 8..15; col = lane&15)
  const int rowbase = (lane >> 4) * 8;
  const int cl      = lane & 15;
  float* Cfb = Cf ? Cf + (long)b1 * sC1 + (long)b2 * sC2 : nullptr;
  u16*   Cbb = Cb ? Cb + (long)b1 * sC1 + (long)b2 * sC2 : nullptr;
  const float* resb = res ? res + (long)b1 * sC1 + (long)b2 * sC2 : nullptr;
#pragma unroll
  for (int f = 0; f < 2; ++f) {
    v8f acc = f ? acc1 : acc0;
    int cg = n0 + nt * 32 + f * 16 + cl;
    if (cg >= Nc) continue;
#pragma unroll
    for (int r = 0; r < 8; ++r) {
      int rg = m0 + mt * 16 + rowbase + r;
      if (rg >= Mr) continue;
      float v = acc[r];
      if (rowscale) v *= rowscale[(long)b1 * sRS1 + rg];
      if (bias)     v += bias[cg];
      if (resb)     v += resb[(long)rg * ldc + cg];
      if (Cfb) Cfb[(long)rg * ldc + cg] = v;
      if (Cbb) Cbb[(long)rg * ldc + cg] = f2bf(v);
    }
  }
}

// ---------------- fused feat GEMM + row softmax ----------------
// tf = softmax(X @ W + b) over M=512.  Writes tf (row-major, optional) and tf^T.
template <int WRITE_RM>
__global__ __launch_bounds__(256) void feat_softmax_kernel(
    const u16* __restrict__ X,     // (B,N,D) bf16
    const u16* __restrict__ W,     // (D,M)   bf16
    const float* __restrict__ bias,// (M)
    u16* __restrict__ tf_rm,       // (B,N,M) bf16 or null
    u16* __restrict__ tfT)         // (B,M,N) bf16
{
  __shared__ alignas(16) char smem[1024 + 32768 + 1024 + 128];
  u16*   ldsA = (u16*)smem;                           // 16x32 A tile (row-major)
  u16*   ldsB = (u16*)(smem + 1024);                  // col-major [512 n][32 k]
  float* sval = (float*)(smem + 1024);                // overlays ldsB: 16x512 f32
  float* red  = (float*)(smem + 1024 + 32768);        // 16x16
  float* rowv = (float*)(smem + 1024 + 32768 + 1024); // 32

  const int t    = threadIdx.x;
  const int lane = t & 31;
  const int wave = t >> 5;
  const int b    = blockIdx.y;
  const int n0   = blockIdx.x * 16;
  const u16* Xrow = X + ((long)b * Nn + n0) * Dd;

  v8f acc[4] = {};                                    // wave owns cols [wave*64,+64)

  for (int kk = 0; kk < Dd; kk += 32) {
    // ---- stage A 16x32 ----
#if HAVE_TDM
    if (wave == 0) {
      tdm_load_tile(ldsA, Xrow + kk, Dd - kk, 16, Dd, 16);
      __builtin_amdgcn_s_wait_tensorcnt(0);
    }
#else
    if (t < 64) {
      int idx = t * 8, row = idx >> 5, col = idx & 31;
      *(uint4*)&ldsA[idx] = *(const uint4*)&Xrow[(long)row * Dd + kk + col];
    }
#endif
    // ---- stage W rows kk..kk+31 into col-major [n][32]: pack k-pairs ----
    {
      int k   = (t >> 4) * 2;        // 0,2,..,30
      int nc0 = (t & 15) * 32;       // 32 columns per thread
      const u16* s0 = &W[(long)(kk + k) * Mm];
      const u16* s1 = &W[(long)(kk + k + 1) * Mm];
#pragma unroll
      for (int g = 0; g < 4; ++g) {
        int n = nc0 + g * 8;
        alignas(16) u16 r0[8], r1[8];
        *(uint4*)r0 = *(const uint4*)&s0[n];
        *(uint4*)r1 = *(const uint4*)&s1[n];
#pragma unroll
        for (int j = 0; j < 8; ++j)
          *(u32*)&ldsB[(n + j) * 32 + k] = (u32)r0[j] | ((u32)r1[j] << 16);
      }
    }
    __syncthreads();
    v16bf a = load_frag_a(ldsA, lane);
#pragma unroll
    for (int f = 0; f < 4; ++f) {
      v16bf bf = load_frag_b(&ldsB[(wave * 64 + f * 16) * 32], lane);
      acc[f] = wmma_bf16(a, bf, acc[f]);
    }
    __syncthreads();
  }

  { // dump accumulators (+bias) into sval (ldsB dead now)
    const int rowbase = (lane >> 4) * 8, cl = lane & 15;
#pragma unroll
    for (int f = 0; f < 4; ++f) {
      int c = wave * 64 + f * 16 + cl;
#pragma unroll
      for (int r = 0; r < 8; ++r)
        sval[(rowbase + r) * 512 + c] = acc[f][r] + bias[c];
    }
  }
  __syncthreads();

  const int tr = t >> 4, tc = t & 15;                 // 16 rows x 16 partials
  { float mx = -3.4e38f;
    for (int c = tc * 32; c < tc * 32 + 32; ++c) mx = fmaxf(mx, sval[tr * 512 + c]);
    red[tr * 16 + tc] = mx; }
  __syncthreads();
  if (tc == 0) { float mx = red[tr * 16];
    for (int j = 1; j < 16; ++j) mx = fmaxf(mx, red[tr * 16 + j]);
    rowv[tr] = mx; }
  __syncthreads();
  { float s = 0.f, mx = rowv[tr];
    for (int c = tc * 32; c < tc * 32 + 32; ++c) {
      float e = __expf(sval[tr * 512 + c] - mx);
      sval[tr * 512 + c] = e; s += e; }
    red[tr * 16 + tc] = s; }
  __syncthreads();
  if (tc == 0) { float s = 0.f;
    for (int j = 0; j < 16; ++j) s += red[tr * 16 + j];
    rowv[16 + tr] = 1.0f / s; }
  __syncthreads();

#pragma unroll
  for (int cc = 0; cc < 2; ++cc) {                    // thread owns cols {t, t+256}
    int c = t + cc * 256;
    alignas(16) u16 tmp[16];
#pragma unroll
    for (int r = 0; r < 16; ++r) {
      float v = sval[r * 512 + c] * rowv[16 + r];
      tmp[r] = f2bf(v);
      if (WRITE_RM) tf_rm[((long)b * Nn + n0 + r) * Mm + c] = tmp[r];
    }
    long tb = ((long)b * Mm + c) * Nn + n0;
    *(uint4*)&tfT[tb]     = *(uint4*)&tmp[0];
    *(uint4*)&tfT[tb + 8] = *(uint4*)&tmp[8];
  }
}

// ---------------- deterministic colsum -> 1/(sum+eps) ----------------
__global__ __launch_bounds__(256) void colsum_inv_kernel(const u16* __restrict__ tfT,
                                                         float* __restrict__ inv)
{
  __shared__ float red[256];
  int m = blockIdx.x, b = blockIdx.y, t = threadIdx.x;
  const u16* row = tfT + ((long)b * Mm + m) * Nn;
  float s = 0.f;
  for (int n = t; n < Nn; n += 256) s += bf2f(row[n]);
  red[t] = s; __syncthreads();
  for (int off = 128; off > 0; off >>= 1) {
    if (t < off) red[t] += red[t + off];
    __syncthreads();
  }
  if (t == 0) inv[b * Mm + m] = 1.0f / (red[0] + 1e-8f);
}

// ---------------- xq/xk projection: out[m,p] = inv[m] * sum_n tfT[m,n]*x[n,p] ----
__global__ __launch_bounds__(256) void vec_proj_kernel(
    const u16* __restrict__ tfT, const float* __restrict__ x,
    const float* __restrict__ inv, float* __restrict__ out, long outBatchStride)
{
  __shared__ float red[3][256];
  int m = blockIdx.x, b = blockIdx.y, t = threadIdx.x;
  const u16* row  = tfT + ((long)b * Mm + m) * Nn;
  const float* xb = x + (long)b * Nn * 3;
  float s0 = 0, s1 = 0, s2 = 0;
  for (int n = t; n < Nn; n += 256) {
    float w = bf2f(row[n]);
    s0 += w * xb[n * 3 + 0]; s1 += w * xb[n * 3 + 1]; s2 += w * xb[n * 3 + 2];
  }
  red[0][t] = s0; red[1][t] = s1; red[2][t] = s2; __syncthreads();
  for (int off = 128; off > 0; off >>= 1) {
    if (t < off) { red[0][t] += red[0][t+off]; red[1][t] += red[1][t+off]; red[2][t] += red[2][t+off]; }
    __syncthreads();
  }
  if (t == 0) {
    float iv = inv[b * Mm + m];
    float* o = out + (long)b * outBatchStride + (long)m * 3;
    o[0] = red[0][0] * iv; o[1] = red[1][0] * iv; o[2] = red[2][0] * iv;
  }
}

// ---------------- LayerNorm over D=384 ----------------
__global__ __launch_bounds__(128) void ln_kernel(
    const float* __restrict__ in, const float* __restrict__ g, const float* __restrict__ bt,
    float* __restrict__ outF, u16* __restrict__ outB)
{
  __shared__ float r1[128], r2[128];
  int row = blockIdx.x, t = threadIdx.x;
  const float* p = in + (long)row * Dd;
  float s = 0, ss = 0;
  for (int c = t; c < Dd; c += 128) { float v = p[c]; s += v; ss += v * v; }
  r1[t] = s; r2[t] = ss; __syncthreads();
  for (int off = 64; off > 0; off >>= 1) {
    if (t < off) { r1[t] += r1[t + off]; r2[t] += r2[t + off]; }
    __syncthreads();
  }
  float mean = r1[0] / Dd;
  float var  = r2[0] / Dd - mean * mean;
  float rstd = rsqrtf(var + 1e-5f);
  for (int c = t; c < Dd; c += 128) {
    float v = (p[c] - mean) * rstd * g[c] + bt[c];
    if (outF) outF[(long)row * Dd + c] = v;
    if (outB) outB[(long)row * Dd + c] = f2bf(v);
  }
}

// ---------------- RoPE (f32 in -> bf16 out) ----------------
__global__ void rope_kernel(const float* __restrict__ tin, const float* __restrict__ coords,
                            u16* __restrict__ outB, int S)
{
  long idx = (long)blockIdx.x * blockDim.x + threadIdx.x;
  long total = (long)Bb * S * Hh * 24;
  if (idx >= total) return;
  int j = idx % 24; long r = idx / 24;
  int h = r % Hh; r /= Hh;
  int s = r % S;  int b = (int)(r / S);
  int p = j >> 3, f = j & 7;
  float fr  = __expf(MINLF + (MAXLF - MINLF) * (float)f / (float)(NFf - 1));
  float ang = coords[((long)b * S + s) * 3 + p] * fr;
  float cs, sn; __sincosf(ang, &sn, &cs);
  long base = ((long)b * S + s) * 384 + h * 48 + 2 * j;
  float t0 = tin[base], t1 = tin[base + 1];
  outB[base]     = f2bf(t0 * cs - t1 * sn);
  outB[base + 1] = f2bf(t0 * sn + t1 * cs);
}

// ---------------- attention softmax over 512 keys ----------------
__global__ __launch_bounds__(256) void attn_softmax_kernel(
    const float* __restrict__ scores, u16* __restrict__ attn)
{
  __shared__ float red[256];
  long row = blockIdx.x;
  const float* p = scores + row * Mm;
  u16* o = attn + row * Mm;
  int t = threadIdx.x;
  const float scale = 0.14433756729740643f;  // 1/sqrt(48)
  float v0 = p[t] * scale, v1 = p[t + 256] * scale;
  red[t] = fmaxf(v0, v1); __syncthreads();
  for (int off = 128; off > 0; off >>= 1) {
    if (t < off) red[t] = fmaxf(red[t], red[t + off]);
    __syncthreads();
  }
  float mx = red[0]; __syncthreads();
  float e0 = __expf(v0 - mx), e1 = __expf(v1 - mx);
  red[t] = e0 + e1; __syncthreads();
  for (int off = 128; off > 0; off >>= 1) {
    if (t < off) red[t] += red[t + off];
    __syncthreads();
  }
  float inv = 1.0f / red[0];
  o[t] = f2bf(e0 * inv); o[t + 256] = f2bf(e1 * inv);
}

// ---------------- misc elementwise ----------------
__global__ void silu_mul_kernel(const float* G, const float* U, u16* O, long n) {
  long i = (long)blockIdx.x * blockDim.x + threadIdx.x;
  if (i < n) { float g = G[i]; float a = g / (1.f + __expf(-g)); O[i] = f2bf(a * U[i]); }
}
__global__ void cvt_bf16_kernel(const float* s, u16* d, long n) {
  long i = (long)blockIdx.x * blockDim.x + threadIdx.x;
  if (i < n) d[i] = f2bf(s[i]);
}
__global__ void copy_batched(const float* src, long sStride, long sOff,
                             float* dst, long dStride, long dOff, int perBatch, int nb) {
  long i = (long)blockIdx.x * blockDim.x + threadIdx.x;
  if (i >= (long)nb * perBatch) return;
  int b = (int)(i / perBatch); long r = i % perBatch;
  dst[(long)b * dStride + dOff + r] = src[(long)b * sStride + sOff + r];
}

// ================= host orchestration =================
extern "C" void kernel_launch(void* const* d_in, const int* in_sizes, int n_in,
                              void* d_out, int out_size, void* d_ws, size_t ws_size,
                              hipStream_t stream)
{
  (void)in_sizes; (void)n_in; (void)out_size; (void)ws_size;
  const float* qx         = (const float*)d_in[0];
  const float* kvx        = (const float*)d_in[1];
  const float* x          = (const float*)d_in[2];
  const float* reg_tok_qx = (const float*)d_in[3];
  const float* reg_tok_x  = (const float*)d_in[4];
  const float* subq_w     = (const float*)d_in[5];
  const float* subq_b     = (const float*)d_in[6];
  const float* subkv_w    = (const float*)d_in[7];
  const float* subkv_b    = (const float*)d_in[8];
  const float* nq_g       = (const float*)d_in[9];
  const float* nq_b       = (const float*)d_in[10];
  const float* nkv_g      = (const float*)d_in[11];
  const float* nkv_b      = (const float*)d_in[12];
  const float* a0_wq      = (const float*)d_in[13];
  const float* a0_bq      = (const float*)d_in[14];
  const float* a0_wk      = (const float*)d_in[15];
  const float* a0_bk      = (const float*)d_in[16];
  const float* a0_wv      = (const float*)d_in[17];
  const float* a0_bv      = (const float*)d_in[18];
  const float* a0_wo      = (const float*)d_in[19];
  const float* a0_bo      = (const float*)d_in[20];
  const float* out_g      = (const float*)d_in[21];
  const float* out_b      = (const float*)d_in[22];
  const float* L_wq       = (const float*)d_in[23];
  const float* L_bq       = (const float*)d_in[24];
  const float* L_wk       = (const float*)d_in[25];
  const float* L_bk       = (const float*)d_in[26];
  const float* L_wv       = (const float*)d_in[27];
  const float* L_bv       = (const float*)d_in[28];
  const float* L_wo       = (const float*)d_in[29];
  const float* L_bo       = (const float*)d_in[30];
  const float* L_ang      = (const float*)d_in[31];
  const float* L_anb      = (const float*)d_in[32];
  const float* L_fng      = (const float*)d_in[33];
  const float* L_fnb      = (const float*)d_in[34];
  const float* L_wg       = (const float*)d_in[35];
  const float* L_bg       = (const float*)d_in[36];
  const float* L_wu       = (const float*)d_in[37];
  const float* L_bu       = (const float*)d_in[38];
  const float* L_wd       = (const float*)d_in[39];
  const float* L_bd       = (const float*)d_in[40];
  float* outF = (float*)d_out;

  // ---- bump allocator over workspace ----
  char* base = (char*)d_ws; size_t off = 0;
  auto alloc = [&](size_t bytes) -> void* {
    size_t ab = (bytes + 255) & ~(size_t)255;
    void* p = base + off; off += ab; return p;
  };
  const long BND = (long)Bb * Nn * Dd;
  u16* qxbf    = (u16*)alloc(BND * 2);
  u16* kvxbf   = (u16*)alloc(BND * 2);
  u16* wqs_bf  = (u16*)alloc((long)Dd * Mm * 2);
  u16* wkvs_bf = (u16*)alloc((long)Dd * Mm * 2);
  u16* a0wq_bf = (u16*)alloc((long)Dd * 384 * 2);
  u16* a0wk_bf = (u16*)alloc((long)Dd * 384 * 2);
  u16* a0wv_bf = (u16*)alloc((long)Dd * 384 * 2);
  u16* a0wo_bf = (u16*)alloc((long)384 * Dd * 2);
  u16* Lwq_bf  = (u16*)alloc((long)Ll * Dd * 384 * 2);
  u16* Lwk_bf  = (u16*)alloc((long)Ll * Dd * 384 * 2);
  u16* Lwv_bf  = (u16*)alloc((long)Ll * Dd * 384 * 2);
  u16* Lwo_bf  = (u16*)alloc((long)Ll * 384 * Dd * 2);
  u16* Lwg_bf  = (u16*)alloc((long)Ll * Dd * FHh * 2);
  u16* Lwu_bf  = (u16*)alloc((long)Ll * Dd * FHh * 2);
  u16* Lwd_bf  = (u16*)alloc((long)Ll * FHh * Dd * 2);
  u16* tfq_rm  = (u16*)alloc((long)Bb * Nn * Mm * 2);
  u16* tfqT    = (u16*)alloc((long)Bb * Mm * Nn * 2);
  u16* tfkvT   = (u16*)alloc((long)Bb * Mm * Nn * 2);
  float* invq  = (float*)alloc((long)Bb * Mm * 4);
  float* invkv = (float*)alloc((long)Bb * Mm * 4);
  float* q_full= (float*)alloc((long)Bb * SQ * Dd * 4);
  float* kvb   = (float*)alloc((long)Bb * Mm * Dd * 4);
  float* xq_f  = (float*)alloc((long)Bb * SQ * 3 * 4);
  float* xk_p  = (float*)alloc((long)Bb * Mm * 3 * 4);
  u16* qln     = (u16*)alloc((long)Bb * SQ * Dd * 2);
  u16* kvln    = (u16*)alloc((long)Bb * Mm * Dd * 2);
  float* y     = (float*)alloc((long)Bb * SQ * Dd * 4);
  float* qh    = (float*)alloc((long)Bb * SQ * 384 * 4);
  float* kh    = (float*)alloc((long)Bb * Mm * 384 * 4);
  u16* vbf     = (u16*)alloc((long)Bb * Mm * 384 * 2);
  u16* qbf     = (u16*)alloc((long)Bb * SQ * 384 * 2);
  u16* kbf     = (u16*)alloc((long)Bb * Mm * 384 * 2);
  float* scores= (float*)alloc((long)Bb * Hh * SQ * Mm * 4);
  u16* attnbf  = (u16*)alloc((long)Bb * Hh * SQ * Mm * 2);
  u16* ofbf    = (u16*)alloc((long)Bb * SQ * 384 * 2);
  u16* hln     = (u16*)alloc((long)Bb * SQ * Dd * 2);
  float* Gf    = (float*)alloc((long)Bb * SQ * FHh * 4);
  float* Uf    = (float*)alloc((long)Bb * SQ * FHh * 4);
  u16* hact    = (u16*)alloc((long)Bb * SQ * FHh * 2);
  u16* qnb     = (u16*)alloc((long)Bb * SQ * Dd * 2);
  float* ylnF  = (float*)alloc((long)Bb * SQ * Dd * 4);
  u16* ylnB    = (u16*)alloc((long)Bb * SQ * Dd * 2);

  auto cvt = [&](const float* s, u16* d, long n) {
    cvt_bf16_kernel<<<dim3((unsigned)((n + 255) / 256)), 256, 0, stream>>>(s, d, n);
  };
  auto gemm = [&](int BT, const u16* A, long sA1, long sA2, int lda,
                  const u16* Bm, long sB1, long sB2, int ldb,
                  float* Cf, u16* Cb, long sC1, long sC2, int ldc,
                  const float* bias, const float* rowscale, long sRS1,
                  const float* res, int Mr, int Nc, int K, int nb1, int nb2) {
    dim3 g((Mr + 63) / 64, (Nc + 63) / 64, nb1 * nb2);
    if (BT)
      wmma_gemm_kernel<1><<<g, 256, 0, stream>>>(A, sA1, sA2, lda, Bm, sB1, sB2, ldb,
          Cf, Cb, sC1, sC2, ldc, bias, rowscale, sRS1, res, Mr, Nc, K, nb2);
    else
      wmma_gemm_kernel<0><<<g, 256, 0, stream>>>(A, sA1, sA2, lda, Bm, sB1, sB2, ldb,
          Cf, Cb, sC1, sC2, ldc, bias, rowscale, sRS1, res, Mr, Nc, K, nb2);
  };

  // ---- bf16 copies of streaming inputs + all weights ----
  cvt(qx,  qxbf,  BND);                 cvt(kvx, kvxbf, BND);
  cvt(subq_w,  wqs_bf,  (long)Dd * Mm); cvt(subkv_w, wkvs_bf, (long)Dd * Mm);
  cvt(a0_wq, a0wq_bf, (long)Dd * 384);  cvt(a0_wk, a0wk_bf, (long)Dd * 384);
  cvt(a0_wv, a0wv_bf, (long)Dd * 384);  cvt(a0_wo, a0wo_bf, (long)384 * Dd);
  cvt(L_wq, Lwq_bf, (long)Ll * Dd * 384); cvt(L_wk, Lwk_bf, (long)Ll * Dd * 384);
  cvt(L_wv, Lwv_bf, (long)Ll * Dd * 384); cvt(L_wo, Lwo_bf, (long)Ll * 384 * Dd);
  cvt(L_wg, Lwg_bf, (long)Ll * Dd * FHh); cvt(L_wu, Lwu_bf, (long)Ll * Dd * FHh);
  cvt(L_wd, Lwd_bf, (long)Ll * FHh * Dd);

  // ---- subspace generators (fused GEMM + softmax) ----
  feat_softmax_kernel<1><<<dim3(Nn / 16, Bb), 256, 0, stream>>>(qxbf,  wqs_bf,  subq_b,  tfq_rm, tfqT);
  feat_softmax_kernel<0><<<dim3(Nn / 16, Bb), 256, 0, stream>>>(kvxbf, wkvs_bf, subkv_b, nullptr, tfkvT);
  colsum_inv_kernel<<<dim3(Mm, Bb), 256, 0, stream>>>(tfqT,  invq);
  colsum_inv_kernel<<<dim3(Mm, Bb), 256, 0, stream>>>(tfkvT, invkv);

  // ---- slice projections: tf^T @ X * inv ----
  gemm(0, tfqT, (long)Mm * Nn, 0, Nn, qxbf, BND, 0, Dd,
       q_full, nullptr, (long)SQ * Dd, 0, Dd, nullptr, invq, Mm, nullptr,
       Mm, Dd, Nn, Bb, 1);
  gemm(0, tfkvT, (long)Mm * Nn, 0, Nn, kvxbf, BND, 0, Dd,
       kvb, nullptr, (long)Mm * Dd, 0, Dd, nullptr, invkv, Mm, nullptr,
       Mm, Dd, Nn, Bb, 1);
  copy_batched<<<(Bb * Rr * Dd + 255) / 256, 256, 0, stream>>>(
      reg_tok_qx, (long)Rr * Dd, 0, q_full, (long)SQ * Dd, (long)Mm * Dd, Rr * Dd, Bb);
  vec_proj_kernel<<<dim3(Mm, Bb), 256, 0, stream>>>(tfqT,  x, invq,  xq_f, (long)SQ * 3);
  vec_proj_kernel<<<dim3(Mm, Bb), 256, 0, stream>>>(tfkvT, x, invkv, xk_p, (long)Mm * 3);
  copy_batched<<<(Bb * Rr * 3 + 255) / 256, 256, 0, stream>>>(
      reg_tok_x, (long)Rr * 3, 0, xq_f, (long)SQ * 3, (long)Mm * 3, Rr * 3, Bb);

  // ---- pre-attention LayerNorms ----
  ln_kernel<<<Bb * SQ, 128, 0, stream>>>(q_full, nq_g,  nq_b,  nullptr, qln);
  ln_kernel<<<Bb * Mm, 128, 0, stream>>>(kvb,    nkv_g, nkv_b, nullptr, kvln);

  // ---- MHA helper ----
  auto run_mha = [&](const u16* qIn, const u16* wq, const float* bq2,
                     const u16* wk, const float* bk2, const u16* wv, const float* bv2,
                     const u16* wo, const float* bo2, const float* res, float* yout) {
    gemm(0, qIn,  (long)SQ * Dd, 0, Dd, wq, 0, 0, 384, qh, nullptr, (long)SQ * 384, 0, 384,
         bq2, nullptr, 0, nullptr, SQ, 384, Dd, Bb, 1);
    gemm(0, kvln, (long)Mm * Dd, 0, Dd, wk, 0, 0, 384, kh, nullptr, (long)Mm * 384, 0, 384,
         bk2, nullptr, 0, nullptr, Mm, 384, Dd, Bb, 1);
    gemm(0, kvln, (long)Mm * Dd, 0, Dd, wv, 0, 0, 384, nullptr, vbf, (long)Mm * 384, 0, 384,
         bv2, nullptr, 0, nullptr, Mm, 384, Dd, Bb, 1);
    long tq = (long)Bb * SQ * Hh * 24;
    rope_kernel<<<(tq + 255) / 256, 256, 0, stream>>>(qh, xq_f, qbf, SQ);
    long tk = (long)Bb * Mm * Hh * 24;
    rope_kernel<<<(tk + 255) / 256, 256, 0, stream>>>(kh, xk_p, kbf, Mm);
    // scores[b,h] = q_h (520x48) @ k_h^T (48x512)
    gemm(1, qbf, (long)SQ * 384, 48, 384, kbf, (long)Mm * 384, 48, 384,
         scores, nullptr, (long)Hh * SQ * Mm, (long)SQ * Mm, Mm,
         nullptr, nullptr, 0, nullptr, SQ, Mm, DHh, Bb, Hh);
    attn_softmax_kernel<<<Bb * Hh * SQ, 256, 0, stream>>>(scores, attnbf);
    // o[b,h] = attn (520x512) @ v_h (512x48)
    gemm(0, attnbf, (long)Hh * SQ * Mm, (long)SQ * Mm, Mm, vbf, (long)Mm * 384, 48, 384,
         nullptr, ofbf, (long)SQ * 384, 48, 384,
         nullptr, nullptr, 0, nullptr, SQ, DHh, Mm, Bb, Hh);
    // y = o @ wo + bo + residual
    gemm(0, ofbf, (long)SQ * 384, 0, 384, wo, 0, 0, Dd, yout, nullptr, (long)SQ * Dd, 0, Dd,
         bo2, nullptr, 0, res, SQ, Dd, 384, Bb, 1);
  };

  run_mha(qln, a0wq_bf, a0_bq, a0wk_bf, a0_bk, a0wv_bf, a0_bv, a0wo_bf, a0_bo,
          q_full, y);

  // ---- L transformer layers: SwiGLU FFN then MHA, both residual ----
  for (int i = 0; i < Ll; ++i) {
    ln_kernel<<<Bb * SQ, 128, 0, stream>>>(y, L_fng + i * Dd, L_fnb + i * Dd, nullptr, hln);
    gemm(0, hln, (long)SQ * Dd, 0, Dd, Lwg_bf + (long)i * Dd * FHh, 0, 0, FHh,
         Gf, nullptr, (long)SQ * FHh, 0, FHh, L_bg + i * FHh, nullptr, 0, nullptr,
         SQ, FHh, Dd, Bb, 1);
    gemm(0, hln, (long)SQ * Dd, 0, Dd, Lwu_bf + (long)i * Dd * FHh, 0, 0, FHh,
         Uf, nullptr, (long)SQ * FHh, 0, FHh, L_bu + i * FHh, nullptr, 0, nullptr,
         SQ, FHh, Dd, Bb, 1);
    long nf = (long)Bb * SQ * FHh;
    silu_mul_kernel<<<(nf + 255) / 256, 256, 0, stream>>>(Gf, Uf, hact, nf);
    gemm(0, hact, (long)SQ * FHh, 0, FHh, Lwd_bf + (long)i * FHh * Dd, 0, 0, Dd,
         y, nullptr, (long)SQ * Dd, 0, Dd, L_bd + i * Dd, nullptr, 0, y,
         SQ, Dd, FHh, Bb, 1);
    ln_kernel<<<Bb * SQ, 128, 0, stream>>>(y, L_ang + i * Dd, L_anb + i * Dd, nullptr, qnb);
    run_mha(qnb,
            Lwq_bf + (long)i * Dd * 384, L_bq + i * 384,
            Lwk_bf + (long)i * Dd * 384, L_bk + i * 384,
            Lwv_bf + (long)i * Dd * 384, L_bv + i * 384,
            Lwo_bf + (long)i * 384 * Dd, L_bo + i * Dd,
            y, y);
  }

  // ---- final LN, reconstruction, register-token output ----
  ln_kernel<<<Bb * SQ, 128, 0, stream>>>(y, out_g, out_b, ylnF, ylnB);
  // y_recon = tf (N x M) @ y_ln[:M,:] (M x D)   -> directly into d_out
  gemm(0, tfq_rm, (long)Nn * Mm, 0, Mm, ylnB, (long)SQ * Dd, 0, Dd,
       outF, nullptr, (long)Nn * Dd, 0, Dd, nullptr, nullptr, 0, nullptr,
       Nn, Dd, Mm, Bb, 1);
  copy_batched<<<(Bb * Rr * Dd + 255) / 256, 256, 0, stream>>>(
      ylnF, (long)SQ * Dd, (long)Mm * Dd,
      outF + (size_t)Bb * Nn * Dd, (long)Rr * Dd, 0, Rr * Dd, Bb);
}
